// MultiHeadAttention_62225486184813
// MI455X (gfx1250) — matrix operand order repaired and checked by reading, not compiled, not run on hardware
//
#include <hip/hip_runtime.h>

// ---------------------------------------------------------------------------
// Types for CDNA5 WMMA (wave32): 16x16x32 bf16 -> f32
// ---------------------------------------------------------------------------
typedef __attribute__((ext_vector_type(16))) __bf16 bf16x16;
typedef __attribute__((ext_vector_type(8)))  __bf16 bf16x8;
typedef __attribute__((ext_vector_type(8)))  float  f32x8;

static __device__ __forceinline__ __bf16 f2bf(float f) {
  unsigned u = __builtin_bit_cast(unsigned, f);
  u += 0x7FFFu + ((u >> 16) & 1u);            // round-to-nearest-even
  unsigned short h = (unsigned short)(u >> 16);
  return __builtin_bit_cast(__bf16, h);
}

static __device__ __forceinline__ f32x8 wmma_bf16(bf16x16 a, bf16x16 b, f32x8 c) {
  return __builtin_amdgcn_wmma_f32_16x16x32_bf16(
      /*neg_a=*/false, a, /*neg_b=*/false, b,
      /*c_mod=*/(short)0, c, /*reuse_a=*/false, /*reuse_b=*/false);
}

// A fragment (16x32, row-major source, base at (row0,k0), leading dim ld):
// lanes 0-15 : row=lane,    K = 0..15
// lanes 16-31: row=lane-16, K = 16..31
static __device__ __forceinline__ bf16x16 load_a_frag(const __bf16* base, int ld, int lane) {
  int r  = lane & 15;
  int kk = (lane >> 4) << 4;                  // 0 or 16
  const __bf16* p = base + r * ld + kk;
  bf16x8 lo = *(const bf16x8*)(p);
  bf16x8 hi = *(const bf16x8*)(p + 8);
  return __builtin_shufflevector(lo, hi, 0,1,2,3,4,5,6,7,8,9,10,11,12,13,14,15);
}

// B fragment (32x16) from "K-contiguous" storage T[n][k] (leading dim ld):
// lanes 0-15 : n=lane,    K = 0..15 ; lanes 16-31: n=lane-16, K = 16..31
static __device__ __forceinline__ bf16x16 load_b_frag(const __bf16* base, int ld, int lane) {
  int n  = lane & 15;
  int kk = (lane >> 4) << 4;                  // 0 or 16
  const __bf16* p = base + n * ld + kk;
  return *(const bf16x16*)p;
}

// Low 32 bits of a generic LDS pointer == DS byte offset
static __device__ __forceinline__ unsigned lds_off_u32(const void* p) {
  return (unsigned)(size_t)p;
}

// CDNA5 async copy global->LDS, 16B per lane, ASYNCcnt-tracked (GVS addressing)
static __device__ __forceinline__ void async_copy_b128(unsigned lds_byte, int g_byte_off,
                                                       const void* base) {
  asm volatile("global_load_async_to_lds_b128 %0, %1, %2"
               :: "v"(lds_byte), "v"(g_byte_off), "s"(base)
               : "memory");
}

// ---------------------------------------------------------------------------
// Problem constants
// ---------------------------------------------------------------------------
#define BATCH 4
#define SEQ   2048
#define DMODEL 1024
#define NHEAD 16
#define HDIM  64
#define MTOT  (BATCH * SEQ)                   // 8192
#define KSTEP 32
#define NKITER (DMODEL / KSTEP)               // 32

// ---------------------------------------------------------------------------
// Stage 0a: f32 -> bf16 elementwise
// ---------------------------------------------------------------------------
__global__ void __launch_bounds__(256) cvt_f32_bf16(const float* __restrict__ in,
                                                    __bf16* __restrict__ out, int n) {
  int i = blockIdx.x * 256 + threadIdx.x;
  if (i < n) out[i] = f2bf(in[i]);
}

// ---------------------------------------------------------------------------
// Stage 0b: transpose + convert the 4 weight matrices: T[n][k] = W[k][n]
// ---------------------------------------------------------------------------
__global__ void __launch_bounds__(256) transpose_cvt(
    const float* __restrict__ W0, const float* __restrict__ W1,
    const float* __restrict__ W2, const float* __restrict__ W3,
    __bf16* __restrict__ T0, __bf16* __restrict__ T1,
    __bf16* __restrict__ T2, __bf16* __restrict__ T3) {
  __shared__ float tile[32][33];
  const float* W = (blockIdx.z == 0) ? W0 : (blockIdx.z == 1) ? W1
                  : (blockIdx.z == 2) ? W2 : W3;
  __bf16* T = (blockIdx.z == 0) ? T0 : (blockIdx.z == 1) ? T1
             : (blockIdx.z == 2) ? T2 : T3;
  int bx = blockIdx.x * 32, by = blockIdx.y * 32;
  int tx = threadIdx.x, ty = threadIdx.y;     // 32 x 8
#pragma unroll
  for (int i = 0; i < 32; i += 8)
    tile[ty + i][tx] = W[(by + ty + i) * DMODEL + bx + tx];
  __syncthreads();
#pragma unroll
  for (int i = 0; i < 32; i += 8)
    T[(bx + ty + i) * DMODEL + by + tx] = f2bf(tile[tx][ty + i]);
}

// ---------------------------------------------------------------------------
// Shared GEMM core: block = 8 waves -> 128(M) x 64(N) tile, K loop of 32.
// Weight tile (64x32 bf16 = 4KB) staged to LDS with async b128 copies,
// double-buffered; A fragment register-prefetched one K-step ahead.
// ---------------------------------------------------------------------------
static __device__ __forceinline__ void gemm_core_128x64(
    const __bf16* __restrict__ X, const __bf16* __restrict__ Wt,
    int m0, int n0, __bf16* btile /* [2][64*KSTEP] */, f32x8 acc[4]) {
  const int lane = threadIdx.x & 31;
  const int tid  = threadIdx.x;
  const int brow = tid >> 2, bchunk = tid & 3;   // 64 rows x 4 x 16B chunks
  const unsigned lds_base =
      lds_off_u32(btile) + (unsigned)(brow * KSTEP + bchunk * 8) * 2;
  const int g_base = ((n0 + brow) * DMODEL + bchunk * 8) * 2;  // bytes

  // prologue: stage 0 -> buf0, stage 1 -> buf1
  async_copy_b128(lds_base,        g_base,                 Wt);
  async_copy_b128(lds_base + 4096, g_base + KSTEP * 2,     Wt);

  const __bf16* Arow = X + (size_t)m0 * DMODEL;
  bf16x16 a_cur = load_a_frag(Arow, DMODEL, lane);
  const int half = lane >> 4, nl = lane & 15;

  for (int i = 0; i < NKITER; ++i) {
    const int buf = i & 1;
    bf16x16 a_nxt;
    if (i + 1 < NKITER)
      a_nxt = load_a_frag(Arow + (i + 1) * KSTEP, DMODEL, lane);
    // my stage-i copy is complete when outstanding <= 1 (in-order); last iter: 0
    if (i == NKITER - 1) asm volatile("s_wait_asynccnt 0x0" ::: "memory");
    else                 asm volatile("s_wait_asynccnt 0x1" ::: "memory");
    __syncthreads();                            // everyone's stage-i data landed
    const __bf16* bp = btile + buf * (64 * KSTEP);
#pragma unroll
    for (int t = 0; t < 4; ++t) {
      bf16x16 b = *(const bf16x16*)(bp + (t * 16 + nl) * KSTEP + half * 16);
      acc[t] = wmma_bf16(a_cur, b, acc[t]);
    }
    __syncthreads();                            // all waves done reading buf
    if (i + 2 < NKITER)
      async_copy_b128(lds_base + buf * 4096, g_base + (i + 2) * KSTEP * 2, Wt);
    if (i + 1 < NKITER) a_cur = a_nxt;
  }
}

// ---------------------------------------------------------------------------
// Stage 1: QKV projection.  blockIdx.z selects Q/K/V.
// Q,K stored [b][h][n][64]; V stored transposed [b][h][64][n].
// ---------------------------------------------------------------------------
__global__ void __launch_bounds__(256) qkv_gemm(
    const __bf16* __restrict__ X,
    const __bf16* __restrict__ Wqt, const __bf16* __restrict__ Wkt,
    const __bf16* __restrict__ Wvt,
    __bf16* __restrict__ Q, __bf16* __restrict__ K, __bf16* __restrict__ Vt) {
  __shared__ __align__(16) __bf16 bt[2][64 * KSTEP];
  const int lane = threadIdx.x & 31;
  const int wave = threadIdx.x >> 5;
  const int m0 = blockIdx.x * 128 + wave * 16;
  const int n0 = blockIdx.y * 64;
  const int which = blockIdx.z;
  const __bf16* Wt = (which == 0) ? Wqt : (which == 1) ? Wkt : Wvt;

  f32x8 acc[4] = {};
  gemm_core_128x64(X, Wt, m0, n0, &bt[0][0], acc);

  const int half = lane >> 4, nl = lane & 15;
#pragma unroll
  for (int t = 0; t < 4; ++t) {
#pragma unroll
    for (int v = 0; v < 8; ++v) {
      int m   = m0 + v + half * 8;
      int col = n0 + t * 16 + nl;
      int bb  = m >> 11;                      // / SEQ
      int tok = m & (SEQ - 1);
      int h   = col >> 6;
      int d   = col & (HDIM - 1);
      __bf16 val = f2bf(acc[t][v]);
      if (which == 2)
        Vt[((bb * NHEAD + h) * HDIM + d) * SEQ + tok] = val;
      else if (which == 0)
        Q[((bb * NHEAD + h) * SEQ + tok) * HDIM + d] = val;
      else
        K[((bb * NHEAD + h) * SEQ + tok) * HDIM + d] = val;
    }
  }
}

// ---------------------------------------------------------------------------
// Stage 2: causal flash attention.  One wave per (b, h, 16-query tile).
// K fragments prefetched one 32-key block ahead; V fragments issued right
// after the score WMMAs so loads overlap the softmax VALU work.
// ---------------------------------------------------------------------------
__global__ void __launch_bounds__(32) attn_kernel(
    const __bf16* __restrict__ Q, const __bf16* __restrict__ K,
    const __bf16* __restrict__ Vt, __bf16* __restrict__ ctx) {
  __shared__ __align__(16) __bf16 pbuf[16][40];
  const int lane = threadIdx.x & 31;
  const int q0 = blockIdx.x * 16;
  const int h  = blockIdx.y;
  const int b  = blockIdx.z;
  const __bf16* Qh = Q  + (size_t)(b * NHEAD + h) * SEQ * HDIM;
  const __bf16* Kh = K  + (size_t)(b * NHEAD + h) * SEQ * HDIM;
  const __bf16* Vh = Vt + (size_t)(b * NHEAD + h) * HDIM * SEQ;

  bf16x16 qa0 = load_a_frag(Qh + q0 * HDIM + 0,  HDIM, lane);
  bf16x16 qa1 = load_a_frag(Qh + q0 * HDIM + 32, HDIM, lane);

  f32x8 o[4] = {};
  float mrow[8], lrow[8];
#pragma unroll
  for (int v = 0; v < 8; ++v) { mrow[v] = -3.0e38f; lrow[v] = 0.f; }

  const int half = lane >> 4, nl = lane & 15;
  const float L2E = 1.44269504088896f;
  const int kbmax = (q0 + 15) >> 5;

  // preload K fragments for block 0
  bf16x16 kb00 = load_b_frag(Kh + 0  * HDIM + 0,  HDIM, lane);
  bf16x16 kb01 = load_b_frag(Kh + 0  * HDIM + 32, HDIM, lane);
  bf16x16 kb10 = load_b_frag(Kh + 16 * HDIM + 0,  HDIM, lane);
  bf16x16 kb11 = load_b_frag(Kh + 16 * HDIM + 32, HDIM, lane);

  for (int kb = 0; kb <= kbmax; ++kb) {
    const int key0 = kb * 32;
    // ---- scores
    f32x8 s0 = {}, s1 = {};
    s0 = wmma_bf16(qa0, kb00, s0);
    s0 = wmma_bf16(qa1, kb01, s0);
    s1 = wmma_bf16(qa0, kb10, s1);
    s1 = wmma_bf16(qa1, kb11, s1);
    // ---- issue V loads now (consumed after softmax)
    bf16x16 vb0 = load_b_frag(Vh + 0  * SEQ + key0, SEQ, lane);
    bf16x16 vb1 = load_b_frag(Vh + 16 * SEQ + key0, SEQ, lane);
    bf16x16 vb2 = load_b_frag(Vh + 32 * SEQ + key0, SEQ, lane);
    bf16x16 vb3 = load_b_frag(Vh + 48 * SEQ + key0, SEQ, lane);
    // ---- prefetch next block's K fragments
    if (kb < kbmax) {
      const int nk = key0 + 32;
      kb00 = load_b_frag(Kh + (nk     ) * HDIM + 0,  HDIM, lane);
      kb01 = load_b_frag(Kh + (nk     ) * HDIM + 32, HDIM, lane);
      kb10 = load_b_frag(Kh + (nk + 16) * HDIM + 0,  HDIM, lane);
      kb11 = load_b_frag(Kh + (nk + 16) * HDIM + 32, HDIM, lane);
    }
#pragma unroll
    for (int v = 0; v < 8; ++v) { s0[v] *= 0.125f; s1[v] *= 0.125f; }
    // ---- causal mask (only blocks that cross the diagonal)
    if (key0 + 31 > q0) {
#pragma unroll
      for (int v = 0; v < 8; ++v) {
        int qrow = q0 + v + half * 8;
        if (key0 + nl      > qrow) s0[v] = -3.0e38f;
        if (key0 + 16 + nl > qrow) s1[v] = -3.0e38f;
      }
    }
    // ---- online softmax
    float alpha[8];
#pragma unroll
    for (int v = 0; v < 8; ++v) {
      float t = fmaxf(s0[v], s1[v]);
      t = fmaxf(t, __shfl_xor(t, 1, 32));
      t = fmaxf(t, __shfl_xor(t, 2, 32));
      t = fmaxf(t, __shfl_xor(t, 4, 32));
      t = fmaxf(t, __shfl_xor(t, 8, 32));
      float mnew = fmaxf(mrow[v], t);
      alpha[v] = __builtin_amdgcn_exp2f((mrow[v] - mnew) * L2E);
      mrow[v] = mnew;
    }
#pragma unroll
    for (int v = 0; v < 8; ++v) {
      float p0 = __builtin_amdgcn_exp2f((s0[v] - mrow[v]) * L2E);
      float p1 = __builtin_amdgcn_exp2f((s1[v] - mrow[v]) * L2E);
      s0[v] = p0; s1[v] = p1;
      float t = p0 + p1;
      t += __shfl_xor(t, 1, 32);
      t += __shfl_xor(t, 2, 32);
      t += __shfl_xor(t, 4, 32);
      t += __shfl_xor(t, 8, 32);
      lrow[v] = lrow[v] * alpha[v] + t;
    }
#pragma unroll
    for (int t4 = 0; t4 < 4; ++t4)
#pragma unroll
      for (int v = 0; v < 8; ++v) o[t4][v] *= alpha[v];
    // ---- P: C-layout -> A-layout via LDS
#pragma unroll
    for (int v = 0; v < 8; ++v) {
      pbuf[v + half * 8][nl]      = f2bf(s0[v]);
      pbuf[v + half * 8][16 + nl] = f2bf(s1[v]);
    }
    __syncthreads();
    bf16x16 pa;
    {
      const __bf16* p = &pbuf[nl][half * 16];
      bf16x8 lo = *(const bf16x8*)(p);
      bf16x8 hi = *(const bf16x8*)(p + 8);
      pa = __builtin_shufflevector(lo, hi, 0,1,2,3,4,5,6,7,8,9,10,11,12,13,14,15);
    }
    __syncthreads();
    // ---- ctx += P . V
    o[0] = wmma_bf16(pa, vb0, o[0]);
    o[1] = wmma_bf16(pa, vb1, o[1]);
    o[2] = wmma_bf16(pa, vb2, o[2]);
    o[3] = wmma_bf16(pa, vb3, o[3]);
  }
  // ---- normalize and write ctx
  float inv[8];
#pragma unroll
  for (int v = 0; v < 8; ++v) inv[v] = 1.0f / lrow[v];
#pragma unroll
  for (int t4 = 0; t4 < 4; ++t4)
#pragma unroll
    for (int v = 0; v < 8; ++v) {
      int tok = q0 + v + half * 8;
      int col = h * HDIM + t4 * 16 + nl;
      ctx[(size_t)(b * SEQ + tok) * DMODEL + col] = f2bf(o[t4][v] * inv[v]);
    }
}

// ---------------------------------------------------------------------------
// Stage 3: output projection GEMM + bias, f32 out.
// ---------------------------------------------------------------------------
__global__ void __launch_bounds__(256) out_gemm(
    const __bf16* __restrict__ Cx, const __bf16* __restrict__ Wot,
    const float* __restrict__ bo, float* __restrict__ out) {
  __shared__ __align__(16) __bf16 bt[2][64 * KSTEP];
  const int lane = threadIdx.x & 31;
  const int wave = threadIdx.x >> 5;
  const int m0 = blockIdx.x * 128 + wave * 16;
  const int n0 = blockIdx.y * 64;

  f32x8 acc[4] = {};
  gemm_core_128x64(Cx, Wot, m0, n0, &bt[0][0], acc);

  const int half = lane >> 4, nl = lane & 15;
#pragma unroll
  for (int t = 0; t < 4; ++t) {
    int col = n0 + t * 16 + nl;
    float bias = bo[col];
#pragma unroll
    for (int v = 0; v < 8; ++v) {
      int m = m0 + v + half * 8;
      out[(size_t)m * DMODEL + col] = acc[t][v] + bias;
    }
  }
}

// ---------------------------------------------------------------------------
// Launcher.  Workspace layout (bytes):
//   xb 0..16M | Wqt 16M | Wkt 18M | Wvt 20M | Wot 22M |
//   Q 24M..40M | K 40M..56M | Vt 56M..72M | ctx 72M..88M
// ---------------------------------------------------------------------------
extern "C" void kernel_launch(void* const* d_in, const int* in_sizes, int n_in,
                              void* d_out, int out_size, void* d_ws, size_t ws_size,
                              hipStream_t stream) {
  const float* x  = (const float*)d_in[0];
  const float* Wq = (const float*)d_in[1];
  const float* Wk = (const float*)d_in[2];
  const float* Wv = (const float*)d_in[3];
  const float* Wo = (const float*)d_in[4];
  const float* bo = (const float*)d_in[5];
  float* out = (float*)d_out;

  char* ws = (char*)d_ws;
  const size_t MB = 1u << 20;
  __bf16* xb  = (__bf16*)(ws + 0 * MB);
  __bf16* Wqt = (__bf16*)(ws + 16 * MB);
  __bf16* Wkt = (__bf16*)(ws + 18 * MB);
  __bf16* Wvt = (__bf16*)(ws + 20 * MB);
  __bf16* Wot = (__bf16*)(ws + 22 * MB);
  __bf16* Qb  = (__bf16*)(ws + 24 * MB);
  __bf16* Kb  = (__bf16*)(ws + 40 * MB);
  __bf16* Vtb = (__bf16*)(ws + 56 * MB);
  __bf16* Cxb = (__bf16*)(ws + 72 * MB);

  cvt_f32_bf16<<<(MTOT * DMODEL) / 256, 256, 0, stream>>>(x, xb, MTOT * DMODEL);
  transpose_cvt<<<dim3(32, 32, 4), dim3(32, 8), 0, stream>>>(
      Wq, Wk, Wv, Wo, Wqt, Wkt, Wvt, Wot);
  qkv_gemm<<<dim3(MTOT / 128, DMODEL / 64, 3), 256, 0, stream>>>(
      xb, Wqt, Wkt, Wvt, Qb, Kb, Vtb);
  attn_kernel<<<dim3(SEQ / 16, NHEAD, BATCH), 32, 0, stream>>>(Qb, Kb, Vtb, Cxb);
  out_gemm<<<dim3(MTOT / 128, DMODEL / 64), 256, 0, stream>>>(Cxb, Wot, bo, out);
}
